// FastAttentionBlock_89223650607359
// MI455X (gfx1250) — matrix operand order repaired and checked by reading, not compiled, not run on hardware
//
#include <hip/hip_runtime.h>

// FastAttentionBlock for MI455X (gfx1250, wave32, WMMA).
// Pipeline: proj_qk -> proj_v -> sim_partial (split-K) -> sim_reduce -> context.
// All GEMMs use v_wmma_f32_16x16x32_bf16; intermediates bf16, accum f32.
// All LDS fragment reads are contiguous 32B per lane (2 x ds_load_b128);
// LDS rows padded to 40 bf16 (80B) to stagger banks.
// sim_partial/context use double-buffered CDNA5 async global->LDS staging
// (global_load_async_to_lds_b128 + s_wait_asynccnt) to overlap loads with WMMA.
// Workspace layout (needs >= 290 MiB):
//   [0,64Mi)    Qh   bf16 [B][N][256]   (q / ||q||, pixel-major)
//   [64,128Mi)  Kc   bf16 [B][256][N]   (k/(k/||k||), channel-major)
//   [128,256Mi) Vc   bf16 [B][512][N]   (value, channel-major)
//   [256,288Mi) part f32  [SPLITK][B][256][512]  (deterministic split-K partials)
//   [288,290Mi) simbT bf16 [B][512][256]  (sim transposed: [v][c])

#define B_DIM 8
#define C_IN 512
#define C_KQ 256
#define C_OUT 512
#define NPIX 16384
#define SPLITK 8
#define LDP 40  // padded LDS leading dim (bf16 elems) for 32-wide K tiles

#if __has_builtin(__builtin_amdgcn_global_load_async_to_lds_b128) && \
    __has_builtin(__builtin_amdgcn_s_wait_asynccnt)
#define HAVE_ASYNC 1
#else
#define HAVE_ASYNC 0
#endif

typedef __attribute__((ext_vector_type(16))) __bf16 v16bf;
typedef __attribute__((ext_vector_type(8)))  float  v8f;

#if HAVE_ASYNC
typedef int async_v4i __attribute__((vector_size(16)));
typedef __attribute__((address_space(1))) async_v4i* gbl_v4i_p;
typedef __attribute__((address_space(3))) async_v4i* lds_v4i_p;
#endif

__device__ __forceinline__ __bf16 to_bf16(float f) { return (__bf16)f; }

__device__ __forceinline__ void async_cp16(const __bf16* g, __bf16* l) {
#if HAVE_ASYNC
  __builtin_amdgcn_global_load_async_to_lds_b128((gbl_v4i_p)g, (lds_v4i_p)l, 0, 0);
#else
  *(uint4*)l = *(const uint4*)g;
#endif
}

// s_wait_asynccnt needs an immediate -> dedicated helpers.
__device__ __forceinline__ void async_wait0() {
#if HAVE_ASYNC
  __builtin_amdgcn_s_wait_asynccnt(0);
#endif
}
__device__ __forceinline__ void async_wait2() {
#if HAVE_ASYNC
  __builtin_amdgcn_s_wait_asynccnt(2);  // allow the 2 just-issued prefetch ops
#endif
}

__device__ __forceinline__ v8f vzero8() {
  v8f z;
#pragma unroll
  for (int i = 0; i < 8; ++i) z[i] = 0.0f;
  return z;
}

__device__ __forceinline__ v8f wmma_bf16(v16bf a, v16bf b, v8f c) {
  return __builtin_amdgcn_wmma_f32_16x16x32_bf16(false, a, false, b, (short)0, c,
                                                 false, false);
}

// A fragment (16x32, 16-bit): lanes 0-15 hold K 0-15, lanes 16-31 hold
// K 16-31, row m = lane%16. Tile row-major [m][k], leading dim ldk.
// 16 contiguous elements -> 2 x ds_load_b128.
__device__ __forceinline__ v16bf frag_a_rm(const __bf16* t, int ldk) {
  const int lane = threadIdx.x & 31;
  const int m = lane & 15;
  const int kh = (lane & 16) ? 16 : 0;
  v16bf a;
#pragma unroll
  for (int e = 0; e < 8; ++e) a[e] = t[m * ldk + kh + e];
#pragma unroll
  for (int e = 0; e < 8; ++e) a[8 + e] = t[m * ldk + kh + 8 + e];
  return a;
}

// B fragment (32x16, 16-bit): lanes 0-15 -> K 0-15, lanes 16-31 -> K 16-31,
// col n = lane%16. Tile stored column-per-lane [n][k], leading dim ldk.
// 16 contiguous elements -> 2 x ds_load_b128.
__device__ __forceinline__ v16bf frag_b_cm(const __bf16* t, int ldk) {
  const int lane = threadIdx.x & 31;
  const int n = lane & 15;
  const int kb = (lane & 16) ? 16 : 0;
  v16bf b;
#pragma unroll
  for (int e = 0; e < 16; ++e) b[e] = t[n * ldk + kb + e];
  return b;
}

// ---------------------------------------------------------------------------
// Kernel 1: Q & K projection + bias + norm epilogue. WG = 256 ch x 64 px tile.
// ---------------------------------------------------------------------------
__global__ __launch_bounds__(256)
void proj_qk_kernel(const float* __restrict__ x,
                    const float* __restrict__ Wq, const float* __restrict__ bq,
                    const float* __restrict__ Wk, const float* __restrict__ bk,
                    __bf16* __restrict__ Qh, __bf16* __restrict__ Kc) {
  __shared__ __align__(16) __bf16 sWq[C_KQ * LDP];
  __shared__ __align__(16) __bf16 sWk[C_KQ * LDP];
  __shared__ __align__(16) __bf16 sX[64 * LDP];  // [pixel][channel]
  __shared__ float sbq[C_KQ], sbk[C_KQ];
  __shared__ float pqs[64][4], pks[64][4];
  __shared__ float snq[64], snk[64];

  const int t = threadIdx.x;
  const int wave = t >> 5;
  const int lane = t & 31;
  const int b = blockIdx.y;
  const int p0 = blockIdx.x * 64;
  const int ntile = wave & 3;   // 16-pixel column group
  const int mgrp = wave >> 2;   // channel half: 8 m-tiles each

  sbq[t] = bq[t];
  sbk[t] = bk[t];

  v8f accq[8], acck[8];
#pragma unroll
  for (int i = 0; i < 8; ++i) { accq[i] = vzero8(); acck[i] = vzero8(); }

  for (int k0 = 0; k0 < C_IN; k0 += 32) {
    // Stage Wq/Wk chunk (256x32 f32 -> bf16), packed 8B LDS stores.
#pragma unroll
    for (int i = 0; i < 8; ++i) {
      int q = t + i * 256;
      int m = q >> 3;
      int kq = (q & 7) << 2;
      float4 fq = *(const float4*)(Wq + (size_t)m * C_IN + k0 + kq);
      float4 fk = *(const float4*)(Wk + (size_t)m * C_IN + k0 + kq);
      union { uint2 u; __bf16 h[4]; } wq, wk;
      wq.h[0] = to_bf16(fq.x); wq.h[1] = to_bf16(fq.y);
      wq.h[2] = to_bf16(fq.z); wq.h[3] = to_bf16(fq.w);
      wk.h[0] = to_bf16(fk.x); wk.h[1] = to_bf16(fk.y);
      wk.h[2] = to_bf16(fk.z); wk.h[3] = to_bf16(fk.w);
      *(uint2*)(sWq + m * LDP + kq) = wq.u;
      *(uint2*)(sWk + m * LDP + kq) = wk.u;
    }
    // Stage x chunk transposed -> sX[pixel][channel]; 4 coalesced b32 loads
    // per thread (consecutive pixels across lanes), one 8B LDS store.
#pragma unroll
    for (int i = 0; i < 2; ++i) {
      int g = t + i * 256;        // 0..511
      int p = g & 63;
      int c4 = (g >> 6) << 2;     // 0,4,...,28
      const float* xb = x + ((size_t)b * C_IN + k0 + c4) * NPIX + p0 + p;
      union { uint2 u; __bf16 h[4]; } vx;
      vx.h[0] = to_bf16(xb[0]);
      vx.h[1] = to_bf16(xb[NPIX]);
      vx.h[2] = to_bf16(xb[2 * NPIX]);
      vx.h[3] = to_bf16(xb[3 * NPIX]);
      *(uint2*)(sX + p * LDP + c4) = vx.u;
    }
    __syncthreads();
    v16bf bf = frag_b_cm(sX + (ntile * 16) * LDP, LDP);
#pragma unroll
    for (int mt = 0; mt < 8; ++mt) {
      v16bf aq = frag_a_rm(sWq + (mgrp * 8 + mt) * (16 * LDP), LDP);
      accq[mt] = wmma_bf16(aq, bf, accq[mt]);
    }
#pragma unroll
    for (int mt = 0; mt < 8; ++mt) {
      v16bf ak = frag_a_rm(sWk + (mgrp * 8 + mt) * (16 * LDP), LDP);
      acck[mt] = wmma_bf16(ak, bf, acck[mt]);
    }
    __syncthreads();
  }

  // Epilogue: bias + deterministic per-pixel norm reduction (4 fixed slots).
  const int nloc = ntile * 16 + (lane & 15);
  const int hoff = (lane & 16) ? 8 : 0;
  float ssq = 0.f, ssk = 0.f;
#pragma unroll
  for (int mt = 0; mt < 8; ++mt) {
    const int cb = (mgrp * 8 + mt) * 16 + hoff;
#pragma unroll
    for (int j = 0; j < 8; ++j) {
      float vq = accq[mt][j] + sbq[cb + j];
      float vk = acck[mt][j] + sbk[cb + j];
      accq[mt][j] = vq; acck[mt][j] = vk;
      ssq += vq * vq; ssk += vk * vk;
    }
  }
  const int slot = mgrp * 2 + ((lane >> 4) & 1);
  pqs[nloc][slot] = ssq;
  pks[nloc][slot] = ssk;
  __syncthreads();
  if (t < 64) {
    snq[t] = sqrtf(pqs[t][0] + pqs[t][1] + pqs[t][2] + pqs[t][3]);
    snk[t] = sqrtf(pks[t][0] + pks[t][1] + pks[t][2] + pks[t][3]);
  }
  __syncthreads();
  const float qn = snq[nloc];
  const float kn = snk[nloc];
  const size_t n = (size_t)p0 + nloc;
  __bf16* qdst = Qh + ((size_t)b * NPIX + n) * C_KQ;        // [n][c]
  __bf16* kdst = Kc + (size_t)b * C_KQ * NPIX + n;          // [c][n]
#pragma unroll
  for (int mt = 0; mt < 8; ++mt) {
    const int cb = (mgrp * 8 + mt) * 16 + hoff;
    union { uint4 u; __bf16 h[8]; } uq;
#pragma unroll
    for (int j = 0; j < 8; ++j) {
      uq.h[j] = to_bf16(accq[mt][j] / qn);
      float l2 = acck[mt][j] / kn;              // key_l2 = k/||k||
      kdst[(size_t)(cb + j) * NPIX] = to_bf16(acck[mt][j] / l2);  // faithful
    }
    *(uint4*)(qdst + cb) = uq.u;
  }
}

// ---------------------------------------------------------------------------
// Kernel 2: V projection + bias, stored channel-major Vc[b][v][n].
// ---------------------------------------------------------------------------
__global__ __launch_bounds__(256)
void proj_v_kernel(const float* __restrict__ x, const float* __restrict__ Wv,
                   const float* __restrict__ bv, __bf16* __restrict__ Vc) {
  __shared__ __align__(16) __bf16 sW[C_OUT * LDP];
  __shared__ __align__(16) __bf16 sX[64 * LDP];
  __shared__ float sbv[C_OUT];

  const int t = threadIdx.x;
  const int wave = t >> 5;
  const int lane = t & 31;
  const int b = blockIdx.y;
  const int p0 = blockIdx.x * 64;
  const int ntile = wave & 3;
  const int mgrp = wave >> 2;   // channel half: 16 m-tiles each

  sbv[t] = bv[t];
  sbv[t + 256] = bv[t + 256];

  v8f acc[16];
#pragma unroll
  for (int i = 0; i < 16; ++i) acc[i] = vzero8();

  for (int k0 = 0; k0 < C_IN; k0 += 32) {
#pragma unroll
    for (int i = 0; i < 16; ++i) {
      int q = t + i * 256;
      int m = q >> 3;
      int kq = (q & 7) << 2;
      float4 f = *(const float4*)(Wv + (size_t)m * C_IN + k0 + kq);
      union { uint2 u; __bf16 h[4]; } w;
      w.h[0] = to_bf16(f.x); w.h[1] = to_bf16(f.y);
      w.h[2] = to_bf16(f.z); w.h[3] = to_bf16(f.w);
      *(uint2*)(sW + m * LDP + kq) = w.u;
    }
#pragma unroll
    for (int i = 0; i < 2; ++i) {
      int g = t + i * 256;
      int p = g & 63;
      int c4 = (g >> 6) << 2;
      const float* xb = x + ((size_t)b * C_IN + k0 + c4) * NPIX + p0 + p;
      union { uint2 u; __bf16 h[4]; } vx;
      vx.h[0] = to_bf16(xb[0]);
      vx.h[1] = to_bf16(xb[NPIX]);
      vx.h[2] = to_bf16(xb[2 * NPIX]);
      vx.h[3] = to_bf16(xb[3 * NPIX]);
      *(uint2*)(sX + p * LDP + c4) = vx.u;
    }
    __syncthreads();
    v16bf bf = frag_b_cm(sX + (ntile * 16) * LDP, LDP);
#pragma unroll
    for (int mt = 0; mt < 16; ++mt) {
      v16bf af = frag_a_rm(sW + (mgrp * 16 + mt) * (16 * LDP), LDP);
      acc[mt] = wmma_bf16(af, bf, acc[mt]);
    }
    __syncthreads();
  }

  const int nloc = ntile * 16 + (lane & 15);
  const int hoff = (lane & 16) ? 8 : 0;
  const size_t n = (size_t)p0 + nloc;
  __bf16* dst = Vc + (size_t)b * C_OUT * NPIX + n;   // [v][n]
#pragma unroll
  for (int mt = 0; mt < 16; ++mt) {
    const int cb = (mgrp * 16 + mt) * 16 + hoff;
#pragma unroll
    for (int j = 0; j < 8; ++j)
      dst[(size_t)(cb + j) * NPIX] = to_bf16(acc[mt][j] + sbv[cb + j]);
  }
}

// ---------------------------------------------------------------------------
// Kernel 3: sim partials = Kc[c][n] @ Vc[v][n]^T, split-K over N.
// Double-buffered async global->LDS staging overlaps loads with WMMA.
// ---------------------------------------------------------------------------
__global__ __launch_bounds__(256)
void sim_partial_kernel(const __bf16* __restrict__ Kc, const __bf16* __restrict__ Vc,
                        float* __restrict__ part) {
  __shared__ __align__(16) __bf16 sA[2][64 * LDP];  // [c][n] ping-pong
  __shared__ __align__(16) __bf16 sB[2][64 * LDP];  // [v][n] ping-pong

  const int t = threadIdx.x;
  const int wave = t >> 5;
  const int lane = t & 31;
  const int ct = blockIdx.x & 3;       // 64-wide c tile
  const int vt = blockIdx.x >> 2;      // 64-wide v tile
  const int nchunk = blockIdx.y;       // split-K slice
  const int b = blockIdx.z;
  const int c0 = ct * 64, v0 = vt * 64;
  const int nbase = nchunk * (NPIX / SPLITK);
  const int ntcol = wave & 3;
  const int mrow0 = (wave >> 2) * 2;

  v8f acc[2] = {vzero8(), vzero8()};
  const int r = t >> 2;               // tile row 0..63
  const int e8 = (t & 3) << 3;        // 8-elem (16B) quad within 32-wide row

  const __bf16* gA = Kc + ((size_t)b * C_KQ + c0 + r) * NPIX + nbase + e8;
  const __bf16* gB = Vc + ((size_t)b * C_OUT + v0 + r) * NPIX + nbase + e8;

  auto issue = [&](int buf, int kk0) {
    async_cp16(gA + kk0, &sA[buf][r * LDP + e8]);
    async_cp16(gB + kk0, &sB[buf][r * LDP + e8]);
  };

  const int NCH = NPIX / SPLITK;       // 2048
  issue(0, 0);
  for (int kk0 = 0, it = 0; kk0 < NCH; kk0 += 32, ++it) {
    const int cur = it & 1;
    if (kk0 + 32 < NCH) {
      issue(cur ^ 1, kk0 + 32);        // prefetch next chunk into other buffer
      async_wait2();                   // current chunk resident (in-order)
    } else {
      async_wait0();
    }
    __syncthreads();
    v16bf bf = frag_b_cm(&sB[cur][(ntcol * 16) * LDP], LDP);
#pragma unroll
    for (int i = 0; i < 2; ++i) {
      v16bf af = frag_a_rm(&sA[cur][(mrow0 + i) * (16 * LDP)], LDP);
      acc[i] = wmma_bf16(af, bf, acc[i]);
    }
    __syncthreads();                   // cur buffer free for iteration it+2
  }

  const int v = v0 + ntcol * 16 + (lane & 15);
  const int hoff = (lane & 16) ? 8 : 0;
  float* dst = part + (size_t)nchunk * ((size_t)B_DIM * C_KQ * C_OUT);
#pragma unroll
  for (int i = 0; i < 2; ++i) {
    const int cb = c0 + (mrow0 + i) * 16 + hoff;
#pragma unroll
    for (int j = 0; j < 8; ++j)
      dst[((size_t)b * C_KQ + cb + j) * C_OUT + v] = acc[i][j];
  }
}

// ---------------------------------------------------------------------------
// Kernel 4: fixed-order split-K reduce; emits sim TRANSPOSED: simbT[b][v][c].
// ---------------------------------------------------------------------------
__global__ __launch_bounds__(256)
void sim_reduce_kernel(const float* __restrict__ part, __bf16* __restrict__ simbT) {
  const int i = blockIdx.x * 256 + threadIdx.x;   // linear over [b][c][v]
  const int n = B_DIM * C_KQ * C_OUT;
  if (i < n) {
    float s = 0.f;
#pragma unroll
    for (int j = 0; j < SPLITK; ++j) s += part[(size_t)j * n + i];
    const int b = i / (C_KQ * C_OUT);
    const int rem = i - b * (C_KQ * C_OUT);
    const int c = rem / C_OUT;
    const int v = rem - c * C_OUT;
    simbT[((size_t)b * C_OUT + v) * C_KQ + c] = to_bf16(s);
  }
}

// ---------------------------------------------------------------------------
// Kernel 5: context = Qh[n][c] @ simT[v][c]^T, written as out[b][v][n].
// Same double-buffered async staging.
// ---------------------------------------------------------------------------
__global__ __launch_bounds__(256)
void context_kernel(const __bf16* __restrict__ Qh, const __bf16* __restrict__ simbT,
                    float* __restrict__ out) {
  __shared__ __align__(16) __bf16 sA[2][64 * LDP];  // [n][c] ping-pong
  __shared__ __align__(16) __bf16 sB[2][64 * LDP];  // [v][c] ping-pong

  const int t = threadIdx.x;
  const int wave = t >> 5;
  const int lane = t & 31;
  const int n0 = blockIdx.x * 64;
  const int v0 = blockIdx.y * 64;
  const int b = blockIdx.z;
  const int ntcol = wave & 3;
  const int mrow0 = (wave >> 2) * 2;

  v8f acc[2] = {vzero8(), vzero8()};
  const int r = t >> 2;
  const int e8 = (t & 3) << 3;

  const __bf16* gA = Qh + ((size_t)b * NPIX + n0 + r) * C_KQ + e8;
  const __bf16* gB = simbT + ((size_t)b * C_OUT + v0 + r) * C_KQ + e8;

  auto issue = [&](int buf, int c0) {
    async_cp16(gA + c0, &sA[buf][r * LDP + e8]);
    async_cp16(gB + c0, &sB[buf][r * LDP + e8]);
  };

  issue(0, 0);
  for (int c0 = 0, it = 0; c0 < C_KQ; c0 += 32, ++it) {
    const int cur = it & 1;
    if (c0 + 32 < C_KQ) {
      issue(cur ^ 1, c0 + 32);
      async_wait2();
    } else {
      async_wait0();
    }
    __syncthreads();
    v16bf bf = frag_b_cm(&sB[cur][(ntcol * 16) * LDP], LDP);
#pragma unroll
    for (int i = 0; i < 2; ++i) {
      v16bf af = frag_a_rm(&sA[cur][(mrow0 + i) * (16 * LDP)], LDP);
      acc[i] = wmma_bf16(af, bf, acc[i]);
    }
    __syncthreads();
  }

  const int v = v0 + ntcol * 16 + (lane & 15);
  const int hoff = (lane & 16) ? 8 : 0;
#pragma unroll
  for (int i = 0; i < 2; ++i) {
    const int nb = n0 + (mrow0 + i) * 16 + hoff;
    float* dst = out + ((size_t)b * C_OUT + v) * NPIX + nb;
    float4 lo = {acc[i][0], acc[i][1], acc[i][2], acc[i][3]};
    float4 hi = {acc[i][4], acc[i][5], acc[i][6], acc[i][7]};
    *(float4*)dst = lo;
    *(float4*)(dst + 4) = hi;
  }
}

// ---------------------------------------------------------------------------
extern "C" void kernel_launch(void* const* d_in, const int* in_sizes, int n_in,
                              void* d_out, int out_size, void* d_ws, size_t ws_size,
                              hipStream_t stream) {
  const float* x  = (const float*)d_in[0];
  const float* Wq = (const float*)d_in[1];
  const float* bq = (const float*)d_in[2];
  const float* Wk = (const float*)d_in[3];
  const float* bk = (const float*)d_in[4];
  const float* Wv = (const float*)d_in[5];
  const float* bv = (const float*)d_in[6];
  float* out = (float*)d_out;

  char* ws = (char*)d_ws;
  __bf16* Qh    = (__bf16*)(ws);
  __bf16* Kc    = (__bf16*)(ws + ((size_t)64 << 20));
  __bf16* Vc    = (__bf16*)(ws + ((size_t)128 << 20));
  float*  part  = (float*) (ws + ((size_t)256 << 20));
  __bf16* simbT = (__bf16*)(ws + ((size_t)288 << 20));

  proj_qk_kernel<<<dim3(NPIX / 64, B_DIM), 256, 0, stream>>>(x, Wq, bq, Wk, bk, Qh, Kc);
  proj_v_kernel<<<dim3(NPIX / 64, B_DIM), 256, 0, stream>>>(x, Wv, bv, Vc);
  sim_partial_kernel<<<dim3(32, SPLITK, B_DIM), 256, 0, stream>>>(Kc, Vc, part);
  sim_reduce_kernel<<<(B_DIM * C_KQ * C_OUT) / 256, 256, 0, stream>>>(part, simbT);
  context_kernel<<<dim3(NPIX / 64, C_OUT / 64, B_DIM), 256, 0, stream>>>(Qh, simbT, out);
}
